// get_model_69999376990732
// MI455X (gfx1250) — matrix operand order repaired and checked by reading, not compiled, not run on hardware
//
#include <hip/hip_runtime.h>

// ---------------------------------------------------------------------------
// PointNet++ MSG forward for MI455X (gfx1250, wave32, WMMA).
// Heavy stage (fp1: 65536 x 1065 -> 512 -> 256 -> 32, ~90 GFLOP) runs on
// v_wmma_f32_16x16x32_bf16. BN folded into bf16 weights, pre-swizzled into
// [ks][nt][lane][16] B-fragment order so a wave's fragments for one K-step
// are contiguous (uniform SGPR base + lane offset + immediate offsets).
// 32 points per workgroup (2 M-tiles) so each B fragment feeds 2 WMMAs.
// Small stages (FPS, ball-query grouping, sa1/sa3/fp3) are VALU kernels with
// enough grid-level parallelism that none serially dominates the WMMA stage.
// ---------------------------------------------------------------------------

typedef __bf16 bf16_t;
typedef __attribute__((ext_vector_type(16))) __bf16 v16bf;
typedef __attribute__((ext_vector_type(8)))  __bf16 v8bf;
typedef __attribute__((ext_vector_type(8)))  float  v8f;

#define BB      16
#define NN      4096
#define NPOINT  24
#define NSAMPLE 24
#define BN_EPS  1e-5f

// fp1 GEMM geometry
#define K1PAD 1088   // 1065 padded to 34*32
#define KS1   34
#define C1    512
#define NT1   32     // 512/16
#define KS2   16     // 512/32
#define C2    256
#define NT2   16
#define KSH   8      // 256/32
#define NTH   2
#define CH_OUT 32
#define MPTS  32     // points per workgroup (2 M-tiles)

struct LayerP { const float *W, *beta, *gamma, *mean, *var; };

struct Params {
    const float *xyz;      // [B,6,N]
    const int   *cls;      // [B]
    const float *g;        // [B,512]
    const float *cls_emb;  // [20,32]
    const float *convW;    // [32,256]
    const float *convB;    // [32]
    LayerP fp1a, fp1b;     // 1065->512, 512->256
    LayerP fp3a, fp3b;     // 1536->1024, 1024->512
    LayerP sa1[6];         // 3 branches x 2 layers
    LayerP sa3a, sa3b;     // 515->256, 256->1024
};

struct WSpace {
    int    *fps;     // [B,24]
    float  *nxyz;    // [B,24,3]
    float  *l1pts;   // [B,24,512]
    float  *sa3tmp;  // [B,24,1024]
    float  *l3;      // [B,1024]
    float  *l1b;     // [B,24,512]
    bf16_t *W1s;     // swizzled [KS1][NT1][32][16]
    bf16_t *W2s;     // swizzled [KS2][NT2][32][16]
    bf16_t *Whs;     // swizzled [KSH][NTH][32][16]
    float  *b1;      // [512]
    float  *b2;      // [256]
    float  *bh;      // [32]
};

// ---------------------------------------------------------------------------
// BN-fold + bf16 convert + B-fragment swizzle.
// outW linear index t == (((ks*NT + nt)*32 + lane)*16 + j)
// fragment element: n = nt*16 + (lane&15), k = ks*32 + (lane>>4)*8 + (j&7) + (j>>3)*16
// ---------------------------------------------------------------------------
__global__ void fold_weights_kernel(const float* W, const float* gamma,
                                    const float* beta, const float* mean,
                                    const float* var, int co, int ci,
                                    int ksteps, bf16_t* outW, float* outB) {
    long t = (long)blockIdx.x * blockDim.x + threadIdx.x;
    const int NT = co >> 4;
    long total = (long)NT * ksteps * 512;
    if (t < total) {
        int perKS = NT * 512;
        int ks = (int)(t / perKS);
        int r  = (int)(t % perKS);
        int nt = r >> 9;
        int r2 = r & 511;
        int lane = r2 >> 4;
        int j    = r2 & 15;
        int k    = ks * 32 + (lane >> 4) * 8 + (j & 7) + (j >> 3) * 16;
        int ch   = nt * 16 + (lane & 15);
        float sc = gamma ? gamma[ch] * rsqrtf(var[ch] + BN_EPS) : 1.0f;
        float v  = (k < ci) ? W[(long)ch * ci + k] * sc : 0.0f;
        outW[t] = (bf16_t)v;
    }
    if (t < co) {
        float sc = gamma ? gamma[t] * rsqrtf(var[t] + BN_EPS) : 1.0f;
        float bb = beta ? beta[t] : 0.0f;
        if (mean) bb -= mean[t] * sc;
        outB[t] = bb;
    }
}

// ---------------------------------------------------------------------------
// Farthest point sampling: one block per batch, LDS argmax reduction.
// ---------------------------------------------------------------------------
__global__ void fps_kernel(Params p, WSpace w) {
    const int b   = blockIdx.x;
    const int tid = threadIdx.x;
    __shared__ float dist[NN];
    __shared__ float rmx[256];
    __shared__ int   rix[256];
    __shared__ int   s_far;
    const float* X = p.xyz + (long)b * 6 * NN;
    for (int n = tid; n < NN; n += 256) dist[n] = 1e10f;
    if (tid == 0) s_far = 0;
    __syncthreads();
    for (int it = 0; it < NPOINT; ++it) {
        int far = s_far;
        if (tid == 0) {
            w.fps[b * NPOINT + it] = far;
            w.nxyz[(b * NPOINT + it) * 3 + 0] = X[far];
            w.nxyz[(b * NPOINT + it) * 3 + 1] = X[NN + far];
            w.nxyz[(b * NPOINT + it) * 3 + 2] = X[2 * NN + far];
        }
        float cx = X[far], cy = X[NN + far], cz = X[2 * NN + far];
        float best = -1.0f; int bidx = NN;
        for (int n = tid; n < NN; n += 256) {
            float dx = X[n] - cx, dy = X[NN + n] - cy, dz = X[2 * NN + n] - cz;
            float d = dx * dx + dy * dy + dz * dz;
            float nd = fminf(dist[n], d);
            dist[n] = nd;
            if (nd > best || (nd == best && n < bidx)) { best = nd; bidx = n; }
        }
        rmx[tid] = best; rix[tid] = bidx;
        __syncthreads();
        for (int off = 128; off > 0; off >>= 1) {
            if (tid < off) {
                if (rmx[tid + off] > rmx[tid] ||
                    (rmx[tid + off] == rmx[tid] && rix[tid + off] < rix[tid])) {
                    rmx[tid] = rmx[tid + off]; rix[tid] = rix[tid + off];
                }
            }
            __syncthreads();
        }
        if (tid == 0) s_far = rix[0];
        __syncthreads();
    }
}

// ---------------------------------------------------------------------------
// sa1: ball-query grouping + tiny 2-layer MLP + max pool over samples.
// One block (64 threads) per (batch, centroid, radius-branch).
// ---------------------------------------------------------------------------
__global__ void sa1_kernel(Params p, WSpace w) {
    const int bi  = blockIdx.x;
    const int b   = bi / (NPOINT * 3);
    const int rem = bi % (NPOINT * 3);
    const int s   = rem / 3;
    const int br  = rem % 3;
    const int tid = threadIdx.x;

    const float R2tab[3]  = {0.01f, 0.04f, 0.09f};
    const int   c1tab[3]  = {64, 128, 64};
    const int   c2tab[3]  = {128, 256, 128};
    const int   offtab[3] = {0, 128, 384};
    const float r2 = R2tab[br];
    const int   c1 = c1tab[br], c2 = c2tab[br], off = offtab[br];

    __shared__ int   grp[NSAMPLE];
    __shared__ int   cnts[65];
    __shared__ float cen[3];
    __shared__ float sfeat[NSAMPLE * 9];
    __shared__ float h1s[128];
    __shared__ float mx[256];

    const float* X = p.xyz + (long)b * 6 * NN;
    if (tid < 3) cen[tid] = w.nxyz[(b * NPOINT + s) * 3 + tid];
    for (int c = tid; c < 256; c += 64) mx[c] = 0.0f;
    __syncthreads();

    const int base = tid * 64;
    int cnt = 0;
    for (int k = 0; k < 64; ++k) {
        int n = base + k;
        float dx = X[n] - cen[0], dy = X[NN + n] - cen[1], dz = X[2 * NN + n] - cen[2];
        if (dx * dx + dy * dy + dz * dz <= r2) cnt++;
    }
    cnts[tid] = cnt;
    __syncthreads();
    if (tid == 0) {
        int run = 0;
        for (int i = 0; i < 64; ++i) { int c = cnts[i]; cnts[i] = run; run += c; }
        cnts[64] = run;
    }
    __syncthreads();
    int pos = cnts[tid];
    for (int k = 0; k < 64; ++k) {
        int n = base + k;
        float dx = X[n] - cen[0], dy = X[NN + n] - cen[1], dz = X[2 * NN + n] - cen[2];
        if (dx * dx + dy * dy + dz * dz <= r2) {
            if (pos < NSAMPLE) grp[pos] = n;
            pos++;
        }
    }
    __syncthreads();
    if (tid == 0) {
        int tot = cnts[64]; if (tot > NSAMPLE) tot = NSAMPLE;
        if (tot == 0) { grp[0] = w.fps[b * NPOINT + s]; tot = 1; }
        for (int k = tot; k < NSAMPLE; ++k) grp[k] = grp[0];
    }
    __syncthreads();

    for (int t = tid; t < NSAMPLE * 9; t += 64) {
        int j = t / 9, c = t % 9;
        int n = grp[j];
        float v = (c < 6) ? X[c * NN + n] : (X[(c - 6) * NN + n] - cen[c - 6]);
        sfeat[t] = v;
    }
    __syncthreads();

    const LayerP L1 = p.sa1[br * 2 + 0];
    const LayerP L2 = p.sa1[br * 2 + 1];
    for (int j = 0; j < NSAMPLE; ++j) {
        for (int c = tid; c < c1; c += 64) {
            float acc = 0.0f;
            for (int k = 0; k < 9; ++k) acc += L1.W[c * 9 + k] * sfeat[j * 9 + k];
            float sc = L1.gamma[c] * rsqrtf(L1.var[c] + BN_EPS);
            h1s[c] = fmaxf((acc - L1.mean[c]) * sc + L1.beta[c], 0.0f);
        }
        __syncthreads();
        for (int c = tid; c < c2; c += 64) {
            float acc = 0.0f;
            for (int k = 0; k < c1; ++k) acc += L2.W[c * c1 + k] * h1s[k];
            float sc = L2.gamma[c] * rsqrtf(L2.var[c] + BN_EPS);
            float v  = fmaxf((acc - L2.mean[c]) * sc + L2.beta[c], 0.0f);
            mx[c] = fmaxf(mx[c], v);
        }
        __syncthreads();
    }
    for (int c = tid; c < c2; c += 64)
        w.l1pts[((long)b * NPOINT + s) * 512 + off + c] = mx[c];
}

// ---------------------------------------------------------------------------
// sa3 per point: 515 -> 256 -> 1024 into sa3tmp. One block per (b, s).
// ---------------------------------------------------------------------------
__global__ void sa3_point_kernel(Params p, WSpace w) {
    const int b = blockIdx.x / NPOINT, s = blockIdx.x % NPOINT;
    const int tid = threadIdx.x;
    __shared__ float row[515];
    __shared__ float h1[256];
    for (int c = tid; c < 515; c += 256)
        row[c] = (c < 3) ? w.nxyz[(b * NPOINT + s) * 3 + c]
                         : w.l1pts[((long)b * NPOINT + s) * 512 + (c - 3)];
    __syncthreads();
    {
        int c = tid;
        float acc = 0.0f;
        for (int k = 0; k < 515; ++k) acc += p.sa3a.W[c * 515 + k] * row[k];
        float sc = p.sa3a.gamma[c] * rsqrtf(p.sa3a.var[c] + BN_EPS);
        h1[c] = fmaxf((acc - p.sa3a.mean[c]) * sc + p.sa3a.beta[c], 0.0f);
    }
    __syncthreads();
    for (int c = tid; c < 1024; c += 256) {
        float acc = 0.0f;
        for (int k = 0; k < 256; ++k) acc += p.sa3b.W[c * 256 + k] * h1[k];
        float sc = p.sa3b.gamma[c] * rsqrtf(p.sa3b.var[c] + BN_EPS);
        w.sa3tmp[((long)b * NPOINT + s) * 1024 + c] =
            fmaxf((acc - p.sa3b.mean[c]) * sc + p.sa3b.beta[c], 0.0f);
    }
}

__global__ void sa3_max_kernel(WSpace w) {
    const int b = blockIdx.x, tid = threadIdx.x;
    for (int c = tid; c < 1024; c += 256) {
        float m = 0.0f;
        for (int s = 0; s < NPOINT; ++s)
            m = fmaxf(m, w.sa3tmp[((long)b * NPOINT + s) * 1024 + c]);
        w.l3[b * 1024 + c] = m;
    }
}

// ---------------------------------------------------------------------------
// fp3 per point: cat(l1_pts[512], l3[1024]) -> 1024 -> 512. Block per (b,s).
// ---------------------------------------------------------------------------
__global__ void fp3_kernel(Params p, WSpace w) {
    const int b = blockIdx.x / NPOINT, s = blockIdx.x % NPOINT;
    const int tid = threadIdx.x;
    __shared__ float l3s[1024];
    __shared__ float row[512];
    __shared__ float h1[1024];
    for (int i = tid; i < 1024; i += 256) l3s[i] = w.l3[b * 1024 + i];
    for (int c = tid; c < 512; c += 256)
        row[c] = w.l1pts[((long)b * NPOINT + s) * 512 + c];
    __syncthreads();
    for (int c = tid; c < 1024; c += 256) {
        const float* Wr = p.fp3a.W + (long)c * 1536;
        float acc = 0.0f;
        for (int k = 0; k < 512; ++k) acc += Wr[k] * row[k];
        for (int k = 0; k < 1024; ++k) acc += Wr[512 + k] * l3s[k];
        float sc = p.fp3a.gamma[c] * rsqrtf(p.fp3a.var[c] + BN_EPS);
        h1[c] = fmaxf((acc - p.fp3a.mean[c]) * sc + p.fp3a.beta[c], 0.0f);
    }
    __syncthreads();
    for (int c = tid; c < 512; c += 256) {
        const float* Wr = p.fp3b.W + (long)c * 1024;
        float acc = 0.0f;
        for (int k = 0; k < 1024; ++k) acc += Wr[k] * h1[k];
        float sc = p.fp3b.gamma[c] * rsqrtf(p.fp3b.var[c] + BN_EPS);
        w.l1b[((long)b * NPOINT + s) * 512 + c] =
            fmaxf((acc - p.fp3b.mean[c]) * sc + p.fp3b.beta[c], 0.0f);
    }
}

// ---------------------------------------------------------------------------
// A fragment (16x32 bf16, ISA layout): lane row = lane&15, kb=(lane>>4)*8;
// elems 0..7 = K kb..kb+7, elems 8..15 = K kb+16..kb+23 -> two 16B ds loads.
// ---------------------------------------------------------------------------
__device__ inline v16bf load_afrag(const bf16_t* rowk) {
    v8bf lo = *(const v8bf*)(rowk);
    v8bf hi = *(const v8bf*)(rowk + 16);
    return __builtin_shufflevector(lo, hi, 0, 1, 2, 3, 4, 5, 6, 7,
                                   8, 9, 10, 11, 12, 13, 14, 15);
}

// ---------------------------------------------------------------------------
// fp1 + head, fused. One block = 32 points, 256 threads = 8 wave32.
// Each B fragment feeds 2 WMMAs (2 M-tiles).
// ---------------------------------------------------------------------------
__global__ void fp1_head_kernel(Params p, WSpace w, float* out) {
    const int b    = blockIdx.x >> 7;       // 128 tiles per batch
    const int til  = blockIdx.x & 127;
    const int n0   = til * MPTS;
    const int tid  = threadIdx.x;
    const int wv   = tid >> 5;
    const int lane = tid & 31;

    extern __shared__ __align__(32) char smem[];
    bf16_t* XA = (bf16_t*)smem;              // [32][K1PAD]  (69632 B)
    bf16_t* X2 = XA + MPTS * K1PAD;          // [32][512]    (32768 B)
    bf16_t* X3 = XA;                         // [32][256] aliases dead XA
    float*  w3 = (float*)(X2 + MPTS * 512);  // [32][3]
    int*    id3 = (int*)(w3 + MPTS * 3);     // [32][3]

    const float* X = p.xyz + (long)b * 6 * NN;

    // ---- 3-NN inverse-distance weights (one thread per point) ----
    if (tid < MPTS) {
        const int n = n0 + tid;
        const float px = X[n], py = X[NN + n], pz = X[2 * NN + n];
        float d0 = 1e30f, d1 = 1e30f, d2 = 1e30f;
        int i0 = 0, i1 = 0, i2 = 0;
        for (int s = 0; s < NPOINT; ++s) {
            const float* c = w.nxyz + (b * NPOINT + s) * 3;
            float dx = px - c[0], dy = py - c[1], dz = pz - c[2];
            float d = dx * dx + dy * dy + dz * dz;
            if (d < d0)      { d2 = d1; i2 = i1; d1 = d0; i1 = i0; d0 = d; i0 = s; }
            else if (d < d1) { d2 = d1; i2 = i1; d1 = d; i1 = s; }
            else if (d < d2) { d2 = d; i2 = s; }
        }
        float r0 = 1.0f / (d0 + 1e-8f), r1 = 1.0f / (d1 + 1e-8f), r2 = 1.0f / (d2 + 1e-8f);
        float inv = 1.0f / (r0 + r1 + r2);
        w3[tid * 3 + 0] = r0 * inv; w3[tid * 3 + 1] = r1 * inv; w3[tid * 3 + 2] = r2 * inv;
        id3[tid * 3 + 0] = i0; id3[tid * 3 + 1] = i1; id3[tid * 3 + 2] = i2;
    }
    __syncthreads();

    // ---- build activation tile XA[32][1088] (bf16) ----
    const int cls = p.cls[b];
    for (int pp = 0; pp < MPTS; ++pp) {
        const int n = n0 + pp;
        for (int c = tid; c < 553; c += 256) {
            float v;
            if (c < 32)        v = p.cls_emb[cls * 32 + c];
            else if (c < 544)  v = p.g[b * 512 + (c - 32)];
            else if (c < 547)  v = X[(c - 544) * NN + n];
            else               v = X[(c - 547) * NN + n];
            XA[pp * K1PAD + c] = (bf16_t)v;
        }
    }
    {
        const float* lb = w.l1b + (long)b * NPOINT * 512;
        for (int t = tid; t < MPTS * 512; t += 256) {
            int pp = t >> 9, c = t & 511;
            float v = w3[pp * 3 + 0] * lb[id3[pp * 3 + 0] * 512 + c]
                    + w3[pp * 3 + 1] * lb[id3[pp * 3 + 1] * 512 + c]
                    + w3[pp * 3 + 2] * lb[id3[pp * 3 + 2] * 512 + c];
            XA[pp * K1PAD + 553 + c] = (bf16_t)v;
        }
        for (int t = tid; t < MPTS * 23; t += 256) {
            int pp = t / 23, c = 1065 + (t % 23);
            XA[pp * K1PAD + c] = (bf16_t)0.0f;
        }
    }
    __syncthreads();

    const int rowA = lane & 15;
    const int kbA  = (lane >> 4) * 8;
    const int col  = lane & 15;
    const int rb   = (lane >> 4) * 8;

    // ---- layer 1: 1088 -> 512 (wave: 4 N-tiles x 2 M-tiles) ----
    v8f acc[2][4];
    for (int m = 0; m < 2; ++m)
        for (int i = 0; i < 4; ++i)
            for (int j = 0; j < 8; ++j) acc[m][i][j] = 0.0f;
    {
        const bf16_t* ap = XA + rowA * K1PAD + kbA;
        const bf16_t* bp = w.W1s + (size_t)(wv * 4) * 512 + lane * 16;
        for (int ks = 0; ks < KS1; ++ks) {
            __builtin_prefetch(bp + (size_t)NT1 * 512, 0, 1);
            v16bf bfr[4];
#pragma unroll
            for (int i = 0; i < 4; ++i)
                bfr[i] = *(const v16bf*)(bp + i * 512);
            v16bf a0 = load_afrag(ap);
            v16bf a1 = load_afrag(ap + 16 * K1PAD);
#pragma unroll
            for (int i = 0; i < 4; ++i) {
                acc[0][i] = __builtin_amdgcn_wmma_f32_16x16x32_bf16(
                    false, a0, false, bfr[i], (short)0, acc[0][i], false, false);
                acc[1][i] = __builtin_amdgcn_wmma_f32_16x16x32_bf16(
                    false, a1, false, bfr[i], (short)0, acc[1][i], false, false);
            }
            ap += 32;
            bp += (size_t)NT1 * 512;
        }
    }
    for (int i = 0; i < 4; ++i) {
        const int ch = (wv * 4 + i) * 16 + col;
        const float bias = w.b1[ch];
        for (int m = 0; m < 2; ++m)
            for (int j = 0; j < 8; ++j) {
                float v = fmaxf(acc[m][i][j] + bias, 0.0f);
                X2[(m * 16 + rb + j) * 512 + ch] = (bf16_t)v;
            }
    }
    __syncthreads();

    // ---- layer 2: 512 -> 256 (wave: 2 N-tiles x 2 M-tiles) ----
    v8f acc2[2][2];
    for (int m = 0; m < 2; ++m)
        for (int i = 0; i < 2; ++i)
            for (int j = 0; j < 8; ++j) acc2[m][i][j] = 0.0f;
    {
        const bf16_t* ap = X2 + rowA * 512 + kbA;
        const bf16_t* bp = w.W2s + (size_t)(wv * 2) * 512 + lane * 16;
        for (int ks = 0; ks < KS2; ++ks) {
            v16bf bfr[2];
#pragma unroll
            for (int i = 0; i < 2; ++i)
                bfr[i] = *(const v16bf*)(bp + i * 512);
            v16bf a0 = load_afrag(ap);
            v16bf a1 = load_afrag(ap + 16 * 512);
#pragma unroll
            for (int i = 0; i < 2; ++i) {
                acc2[0][i] = __builtin_amdgcn_wmma_f32_16x16x32_bf16(
                    false, a0, false, bfr[i], (short)0, acc2[0][i], false, false);
                acc2[1][i] = __builtin_amdgcn_wmma_f32_16x16x32_bf16(
                    false, a1, false, bfr[i], (short)0, acc2[1][i], false, false);
            }
            ap += 32;
            bp += (size_t)NT2 * 512;
        }
    }
    for (int i = 0; i < 2; ++i) {
        const int ch = (wv * 2 + i) * 16 + col;
        const float bias = w.b2[ch];
        for (int m = 0; m < 2; ++m)
            for (int j = 0; j < 8; ++j) {
                float v = fmaxf(acc2[m][i][j] + bias, 0.0f);
                X3[(m * 16 + rb + j) * 256 + ch] = (bf16_t)v;
            }
    }
    __syncthreads();

    // ---- head: 256 -> 32 (waves 0..3: one (N-tile, M-tile) pair each) ----
    if (wv < 4) {
        const int nt = wv & 1;
        const int mt = wv >> 1;
        v8f acc3;
        for (int j = 0; j < 8; ++j) acc3[j] = 0.0f;
        const bf16_t* ap = X3 + (mt * 16 + rowA) * 256 + kbA;
        const bf16_t* bp = w.Whs + (size_t)nt * 512 + lane * 16;
        for (int ks = 0; ks < KSH; ++ks) {
            v16bf bfr = *(const v16bf*)bp;
            v16bf a = load_afrag(ap);
            acc3 = __builtin_amdgcn_wmma_f32_16x16x32_bf16(
                false, a, false, bfr, (short)0, acc3, false, false);
            ap += 32;
            bp += (size_t)NTH * 512;
        }
        const int ch = nt * 16 + col;
        const float bias = w.bh[ch];
        for (int j = 0; j < 8; ++j)
            out[((long)(b * CH_OUT + ch)) * NN + n0 + mt * 16 + rb + j] = acc3[j] + bias;
    }
}

// ---------------------------------------------------------------------------
// Host side
// ---------------------------------------------------------------------------
extern "C" void kernel_launch(void* const* d_in, const int* in_sizes, int n_in,
                              void* d_out, int out_size, void* d_ws, size_t ws_size,
                              hipStream_t stream) {
    (void)in_sizes; (void)n_in; (void)out_size; (void)ws_size;

    Params P;
    int idx = 0;
    P.xyz     = (const float*)d_in[idx++];
    P.cls     = (const int*)  d_in[idx++];
    P.g       = (const float*)d_in[idx++];
    P.cls_emb = (const float*)d_in[idx++];
    P.convW   = (const float*)d_in[idx++];
    P.convB   = (const float*)d_in[idx++];
    auto mk = [&](LayerP& L) {
        L.W     = (const float*)d_in[idx++];
        L.beta  = (const float*)d_in[idx++];
        L.gamma = (const float*)d_in[idx++];
        L.mean  = (const float*)d_in[idx++];
        L.var   = (const float*)d_in[idx++];
    };
    mk(P.fp1a); mk(P.fp1b);
    mk(P.fp3a); mk(P.fp3b);
    for (int i = 0; i < 6; ++i) mk(P.sa1[i]);
    mk(P.sa3a); mk(P.sa3b);

    // workspace layout (256B aligned partitions)
    char* ws = (char*)d_ws;
    size_t off = 0;
    auto take = [&](size_t bytes) -> char* {
        char* ptr = ws + off;
        off = (off + bytes + 255) & ~(size_t)255;
        return ptr;
    };
    WSpace W;
    W.fps    = (int*)   take((size_t)BB * NPOINT * 4);
    W.nxyz   = (float*) take((size_t)BB * NPOINT * 3 * 4);
    W.l1pts  = (float*) take((size_t)BB * NPOINT * 512 * 4);
    W.sa3tmp = (float*) take((size_t)BB * NPOINT * 1024 * 4);
    W.l3     = (float*) take((size_t)BB * 1024 * 4);
    W.l1b    = (float*) take((size_t)BB * NPOINT * 512 * 4);
    W.W1s    = (bf16_t*)take((size_t)KS1 * NT1 * 512 * 2);
    W.W2s    = (bf16_t*)take((size_t)KS2 * NT2 * 512 * 2);
    W.Whs    = (bf16_t*)take((size_t)KSH * NTH * 512 * 2);
    W.b1     = (float*) take(512 * 4);
    W.b2     = (float*) take(256 * 4);
    W.bh     = (float*) take(32 * 4);

    // fold + swizzle weights for the WMMA stages
    {
        long tot1 = (long)NT1 * KS1 * 512;
        fold_weights_kernel<<<(int)((tot1 + 255) / 256), 256, 0, stream>>>(
            P.fp1a.W, P.fp1a.gamma, P.fp1a.beta, P.fp1a.mean, P.fp1a.var,
            C1, 1065, KS1, W.W1s, W.b1);
        long tot2 = (long)NT2 * KS2 * 512;
        fold_weights_kernel<<<(int)((tot2 + 255) / 256), 256, 0, stream>>>(
            P.fp1b.W, P.fp1b.gamma, P.fp1b.beta, P.fp1b.mean, P.fp1b.var,
            C2, 512, KS2, W.W2s, W.b2);
        long toth = (long)NTH * KSH * 512;
        fold_weights_kernel<<<(int)((toth + 255) / 256), 256, 0, stream>>>(
            P.convW, nullptr, P.convB, nullptr, nullptr,
            CH_OUT, 256, KSH, W.Whs, W.bh);
    }

    fps_kernel<<<BB, 256, 0, stream>>>(P, W);
    sa1_kernel<<<BB * NPOINT * 3, 64, 0, stream>>>(P, W);
    sa3_point_kernel<<<BB * NPOINT, 256, 0, stream>>>(P, W);
    sa3_max_kernel<<<BB, 256, 0, stream>>>(W);
    fp3_kernel<<<BB * NPOINT, 256, 0, stream>>>(P, W);

    const size_t smem = (size_t)(MPTS * K1PAD + MPTS * 512) * sizeof(bf16_t)
                        + MPTS * 3 * sizeof(float) + MPTS * 3 * sizeof(int);
    fp1_head_kernel<<<BB * (NN / MPTS), 256, smem, stream>>>(P, W, (float*)d_out);
}
